// PropNet_36593121362468
// MI455X (gfx1250) — matrix operand order repaired and checked by reading, not compiled, not run on hardware
//
#include <hip/hip_runtime.h>
#include <hip/hip_bf16.h>

typedef __attribute__((ext_vector_type(16))) _Float16 v16h;
typedef __attribute__((ext_vector_type(8)))  float    v8f;
typedef unsigned int u32;

#define DEVI __device__ __forceinline__

#if __has_builtin(__builtin_amdgcn_tensor_load_to_lds) && __has_builtin(__builtin_amdgcn_s_wait_tensorcnt)
#define HAVE_TDM 1
#else
#define HAVE_TDM 0
#endif

#if HAVE_TDM
typedef __attribute__((ext_vector_type(4))) unsigned int v4u32;
typedef __attribute__((ext_vector_type(8))) int v8i32;
typedef __attribute__((ext_vector_type(4))) int v4i32;

// Issue a 2D TDM tile load: tile_d1 rows x tile_d0 2-byte elements, global row
// stride row_stride (elements), packed contiguously into LDS at lds_off.
// D# layout per CDNA5 ISA.
DEVI void tdm_load_2d(u32 lds_off, const void* gptr,
                      u32 tile_d0, u32 tile_d1, u32 row_stride) {
  unsigned long long ga = (unsigned long long)(uintptr_t)gptr;
  v4u32 g0;
  g0[0] = 1u;                                                        // count=1 (user descriptor)
  g0[1] = lds_off;                                                   // lds_addr (bytes)
  g0[2] = (u32)ga;                                                   // global_addr[31:0]
  g0[3] = (u32)((ga >> 32) & 0x01FFFFFFull) | (2u << 30);            // global_addr[56:32], type=2
  u32 td0 = row_stride;                                              // tensor_dim0 (elements)
  u32 td1 = tile_d1;                                                 // tensor_dim1
  v8i32 g1;
  g1[0] = (int)(1u << 16);                                           // mask=0, data_size=1 (2B), no pad
  g1[1] = (int)((td0 & 0xFFFFu) << 16);                              // tensor_dim0[15:0]
  g1[2] = (int)(((td0 >> 16) & 0xFFFFu) | ((td1 & 0xFFFFu) << 16));  // td0[31:16] | td1[15:0]
  g1[3] = (int)(((td1 >> 16) & 0xFFFFu) | ((tile_d0 & 0xFFFFu) << 16)); // td1[31:16] | tile_dim0
  g1[4] = (int)(tile_d1 & 0xFFFFu);                                  // tile_dim1; tile_dim2=0
  g1[5] = (int)row_stride;                                           // tensor_dim0_stride[31:0]
  g1[6] = 0;                                                         // stride hi / dim1_stride lo
  g1[7] = 0;
  v4i32 z4 = {0, 0, 0, 0};
  v8i32 z8 = {0, 0, 0, 0, 0, 0, 0, 0};
  // amdgpu-toolchain (clang-23) 6-arg form: (g0, g1, g2, g3, g4, cpol)
  __builtin_amdgcn_tensor_load_to_lds(g0, g1, z4, z4, z8, 0);
}
#endif

// ---------------- WMMA fragment helpers (CDNA5 wave32 layouts) ----------------
DEVI v16h frag_a(const _Float16* base, int ld) {
  int lane = threadIdx.x & 31;
  const _Float16* p = base + (lane & 15) * ld + ((lane >> 4) << 3);
  v16h a;
#pragma unroll
  for (int i = 0; i < 8; ++i) { a[i] = p[i]; a[8 + i] = p[16 + i]; }
  return a;
}
// B from transposed storage [N][K]: two 8-half runs per lane at k=8h and k=16+8h (2x ds_load_b128)
DEVI v16h frag_bt(const _Float16* base, int ldk) {
  int lane = threadIdx.x & 31;
  const _Float16* p = base + (lane & 15) * ldk + ((lane >> 4) << 3);
  v16h b;
#pragma unroll
  for (int i = 0; i < 8; ++i) { b[i] = p[i]; b[8 + i] = p[16 + i]; }
  return b;
}
DEVI v8f wmma_f16(v16h a, v16h b, v8f c) {
  return __builtin_amdgcn_wmma_f32_16x16x32_f16(false, a, false, b, (short)0, c, false, false);
}
DEVI v8f vzero() { v8f z;
#pragma unroll
  for (int i = 0; i < 8; ++i) z[i] = 0.f; return z; }

// ---------------- weight packing (to transposed f16 [cols][K]) ----------------
__global__ void k_pack_eh_t(const float* __restrict__ src, _Float16* __restrict__ dst, int K) {
  int i = blockIdx.x * 256 + threadIdx.x;
  int total = 2 * 128 * K;
  if (i >= total) return;
  int k = i % K, col = i / K;
  int e = col >> 7, h = col & 127;
  dst[i] = (_Float16)src[(e * K + k) * 128 + h];
}
__global__ void k_cvt_t(const float* __restrict__ src, _Float16* __restrict__ dst, int K, int N) {
  int i = blockIdx.x * 256 + threadIdx.x;
  if (i >= K * N) return;
  int k = i % K, n = i / K;
  dst[i] = (_Float16)src[k * N + n];
}

// ---------------- node encoder (tiny, scalar) ----------------
__global__ __launch_bounds__(128) void k_node_enc(
    const float* __restrict__ node_rep, const float* __restrict__ w, const float* __restrict__ bb,
    const float* __restrict__ g, const float* __restrict__ be, const float* __restrict__ mm, const float* __restrict__ vv,
    float* __restrict__ enc32, _Float16* __restrict__ enc16,
    float* __restrict__ eff32, _Float16* __restrict__ eff16) {
  __shared__ float x[32];
  int row = blockIdx.x, tid = threadIdx.x;
  if (tid < 32) x[tid] = node_rep[row * 32 + tid];
  __syncthreads();
  float a = bb[tid];
#pragma unroll 4
  for (int k = 0; k < 32; ++k) a += x[k] * w[k * 128 + tid];
  a = fmaxf(a, 0.f);
  float s = g[tid] * rsqrtf(vv[tid] + 1e-5f);
  a = a * s + (be[tid] - mm[tid] * s);
  enc32[row * 128 + tid] = a; enc16[row * 128 + tid] = (_Float16)a;
  eff32[row * 128 + tid] = a; eff16[row * 128 + tid] = (_Float16)a;
}

// ---------------- edge encoder: [128x96] @ [96x256] per (b,n) ----------------
__global__ __launch_bounds__(256) void k_edge_enc(
    const float* __restrict__ node_rep, const float* __restrict__ edge_rep, const float* __restrict__ edge_type,
    const float* __restrict__ b1, const float* __restrict__ g, const float* __restrict__ be,
    const float* __restrict__ mm, const float* __restrict__ vv,
    const _Float16* __restrict__ w1t,   // [256][96] f16 (transposed)
    _Float16* __restrict__ eenc) {
  __shared__ _Float16 sA[128 * 96];
  __shared__ _Float16 sB[256 * 96];
  __shared__ float sS[256], sT[256];
  int blk = blockIdx.x, b = blk >> 7, tid = threadIdx.x;

  { float s = g[tid] * rsqrtf(vv[tid] + 1e-5f); sS[tid] = s; sT[tid] = be[tid] - mm[tid] * s; }
  for (int i = tid; i < 256 * 96 / 8; i += 256)
    ((uint4*)sB)[i] = ((const uint4*)w1t)[i];
  const float* repr = node_rep + (size_t)blk * 32;
  for (int i = tid; i < 128 * 32; i += 256) {
    int row = i >> 5, k = i & 31;
    sA[row * 96 + k]      = (_Float16)repr[k];
    sA[row * 96 + 32 + k] = (_Float16)node_rep[((size_t)b * 128 + row) * 32 + k];
    sA[row * 96 + 64 + k] = (_Float16)edge_rep[((size_t)blk * 128 + row) * 32 + k];
  }
  __syncthreads();

  int wave = tid >> 5, rowbase = wave * 16, lane = tid & 31;
  int colb = lane & 15, rhalf = (lane >> 4) * 8;
  v8f acc[16];
#pragma unroll
  for (int i = 0; i < 16; ++i) acc[i] = vzero();
#pragma unroll
  for (int ks = 0; ks < 3; ++ks) {
    v16h a = frag_a(sA + rowbase * 96 + ks * 32, 96);
#pragma unroll
    for (int nt = 0; nt < 16; ++nt)
      acc[nt] = wmma_f16(a, frag_bt(sB + (nt * 16) * 96 + ks * 32, 96), acc[nt]);
  }
#pragma unroll
  for (int ht = 0; ht < 8; ++ht) {
    int h = ht * 16 + colb;
#pragma unroll
    for (int r = 0; r < 8; ++r) {
      int row = rowbase + rhalf + r;
      float v0 = fmaxf(acc[ht][r]     + b1[h],       0.f) * sS[h]       + sT[h];
      float v1 = fmaxf(acc[8 + ht][r] + b1[128 + h], 0.f) * sS[128 + h] + sT[128 + h];
      const float* et = edge_type + ((size_t)blk * 128 + row) * 2;
      eenc[((size_t)blk * 128 + row) * 128 + h] = (_Float16)(v0 * et[0] + v1 * et[1]);
    }
  }
}

// ---------------- edge propagation (fused 2-layer MLP + mask + agg) ----------------
__global__ __launch_bounds__(256) void k_edge_prop(
    const float* __restrict__ edge_type,
    const float* __restrict__ b1, const float* __restrict__ b2,
    const float* __restrict__ g, const float* __restrict__ be,
    const float* __restrict__ mm, const float* __restrict__ vv,
    const _Float16* __restrict__ w1t,  // [256][384] f16 (transposed, full K)
    const _Float16* __restrict__ w2t,  // [2][128][128] f16 (transposed per e)
    const float* __restrict__ neff32, const _Float16* __restrict__ neff16,
    const _Float16* __restrict__ edge_in, _Float16* __restrict__ edge_out,
    float* __restrict__ edge_agg) {
  __shared__ _Float16 sA[128 * 128];   // 32 KB
  __shared__ _Float16 sB[256 * 64];    // 32 KB (weight chunk; ldk=64 for w1, 128 for w2)
  __shared__ _Float16 sH[128 * 256];   // 64 KB (h1)
  __shared__ float sRC[256], sS[256], sT[256], sAgg[128];

  int blk = blockIdx.x, b = blk >> 7, tid = threadIdx.x;
  { float s = g[tid] * rsqrtf(vv[tid] + 1e-5f); sS[tid] = s; sT[tid] = be[tid] - mm[tid] * s; }
  if (tid < 128) sAgg[tid] = 0.f;
  { // rank-1 receiver contribution: rc[col] = sum_{k<128} eff_r[k] * w1t[col][k]
    const float* er = neff32 + (size_t)blk * 128;
    const _Float16* wr = w1t + (size_t)tid * 384;
    float a = 0.f;
    for (int k = 0; k < 128; ++k) a += er[k] * (float)wr[k];
    sRC[tid] = a;
  }

  int wave = tid >> 5, rowbase = wave * 16, lane = tid & 31;
  int colb = lane & 15, rhalf = (lane >> 4) * 8;
  const _Float16* aSrc0 = neff16 + (size_t)b * 128 * 128;     // sender features
  const _Float16* aSrc1 = edge_in + (size_t)blk * 128 * 128;  // edge features
  v8f acc[16];
#pragma unroll
  for (int i = 0; i < 16; ++i) acc[i] = vzero();

  for (int seg = 0; seg < 2; ++seg) {
    const _Float16* src = seg ? aSrc1 : aSrc0;
    for (int half = 0; half < 2; ++half) {
      int koff = 128 + seg * 128 + half * 64;
      __syncthreads();
#if HAVE_TDM
      if (tid < 32) {
        if (half == 0) tdm_load_2d((u32)(uintptr_t)sA, src, 128, 128, 128);
        tdm_load_2d((u32)(uintptr_t)sB, w1t + koff, 64, 256, 384);
        __builtin_amdgcn_s_wait_tensorcnt(0);
      }
#else
      if (half == 0)
        for (int i = tid; i < 2048; i += 256) ((uint4*)sA)[i] = ((const uint4*)src)[i];
      for (int i = tid; i < 2048; i += 256) {
        int col = i >> 3, j = i & 7;
        ((uint4*)(sB + col * 64))[j] = ((const uint4*)(w1t + (size_t)col * 384 + koff))[j];
      }
#endif
      __syncthreads();
#pragma unroll
      for (int ks = 0; ks < 2; ++ks) {
        v16h a = frag_a(sA + rowbase * 128 + half * 64 + ks * 32, 128);
#pragma unroll
        for (int nt = 0; nt < 16; ++nt)
          acc[nt] = wmma_f16(a, frag_bt(sB + (nt * 16) * 64 + ks * 32, 64), acc[nt]);
      }
    }
  }
  // epilogue 1: h1 = relu(acc + rank1 + b1) -> sH
#pragma unroll
  for (int nt = 0; nt < 16; ++nt) {
    int col = nt * 16 + colb;
    float add = sRC[col] + b1[col];
#pragma unroll
    for (int r = 0; r < 8; ++r)
      sH[(rowbase + rhalf + r) * 256 + col] = (_Float16)fmaxf(acc[nt][r] + add, 0.f);
  }
  // GEMM2: per e, h1_e [128x128] @ w2[e] [128x128]
#pragma unroll
  for (int i = 0; i < 16; ++i) acc[i] = vzero();
  for (int e = 0; e < 2; ++e) {
    __syncthreads();
#if HAVE_TDM
    if (tid < 32) {
      tdm_load_2d((u32)(uintptr_t)sB, w2t + (size_t)e * 128 * 128, 128, 128, 128);
      __builtin_amdgcn_s_wait_tensorcnt(0);
    }
#else
    for (int i = tid; i < 2048; i += 256)
      ((uint4*)sB)[i] = ((const uint4*)(w2t + (size_t)e * 128 * 128))[i];
#endif
    __syncthreads();
#pragma unroll
    for (int ks = 0; ks < 4; ++ks) {
      v16h a = frag_a(sH + rowbase * 256 + e * 128 + ks * 32, 256);
#pragma unroll
      for (int ht = 0; ht < 8; ++ht)
        acc[e * 8 + ht] = wmma_f16(a, frag_bt(sB + (ht * 16) * 128 + ks * 32, 128), acc[e * 8 + ht]);
    }
  }
  // epilogue 2: relu, BN, edge_type mask, sum over e; write f16 tile + column sums
#pragma unroll
  for (int ht = 0; ht < 8; ++ht) {
    int h = ht * 16 + colb;
#pragma unroll
    for (int r = 0; r < 8; ++r) {
      int row = rowbase + rhalf + r;
      float v0 = fmaxf(acc[ht][r]     + b2[h],       0.f) * sS[h]       + sT[h];
      float v1 = fmaxf(acc[8 + ht][r] + b2[128 + h], 0.f) * sS[128 + h] + sT[128 + h];
      const float* et = edge_type + ((size_t)blk * 128 + row) * 2;
      float o = v0 * et[0] + v1 * et[1];
      edge_out[((size_t)blk * 128 + row) * 128 + h] = (_Float16)o;
      atomicAdd(&sAgg[h], o);
    }
  }
  __syncthreads();
  if (tid < 128) edge_agg[(size_t)blk * 128 + tid] = sAgg[tid];
}

// ---------------- node propagation (tiny, scalar) ----------------
__global__ __launch_bounds__(128) void k_node_prop(
    const float* __restrict__ nenc, const float* __restrict__ neff_in, const float* __restrict__ eagg,
    const float* __restrict__ w1, const float* __restrict__ b1,
    const float* __restrict__ w2, const float* __restrict__ b2,
    const float* __restrict__ g, const float* __restrict__ be,
    const float* __restrict__ mm, const float* __restrict__ vv,
    float* __restrict__ eff32, _Float16* __restrict__ eff16) {
  __shared__ float x[384];
  __shared__ float h1[128];
  int row = blockIdx.x, tid = threadIdx.x;
  x[tid]       = nenc[row * 128 + tid];
  x[128 + tid] = neff_in[row * 128 + tid];
  x[256 + tid] = eagg[row * 128 + tid];
  __syncthreads();
  float a = b1[tid];
  for (int k = 0; k < 384; ++k) a += x[k] * w1[k * 128 + tid];
  h1[tid] = fmaxf(a, 0.f);
  __syncthreads();
  float a2 = b2[tid];
  for (int k = 0; k < 128; ++k) a2 += h1[k] * w2[k * 128 + tid];
  a2 = fmaxf(a2, 0.f);
  float s = g[tid] * rsqrtf(vv[tid] + 1e-5f);
  a2 = a2 * s + (be[tid] - mm[tid] * s);
  eff32[row * 128 + tid] = a2; eff16[row * 128 + tid] = (_Float16)a2;
}

// ---------------- node predictor (tiny, scalar) ----------------
__global__ __launch_bounds__(128) void k_node_pred(
    const float* __restrict__ neff, const float* __restrict__ nenc,
    const float* __restrict__ w1, const float* __restrict__ b1,
    const float* __restrict__ w2, const float* __restrict__ b2,
    const float* __restrict__ g, const float* __restrict__ be,
    const float* __restrict__ mm, const float* __restrict__ vv,
    float* __restrict__ out) {
  __shared__ float x[256];
  __shared__ float h1[128];
  int row = blockIdx.x, tid = threadIdx.x;
  x[tid]       = neff[row * 128 + tid];
  x[128 + tid] = nenc[row * 128 + tid];
  __syncthreads();
  float a = b1[tid];
  for (int k = 0; k < 256; ++k) a += x[k] * w1[k * 128 + tid];
  a = fmaxf(a, 0.f);
  float s = g[tid] * rsqrtf(vv[tid] + 1e-5f);
  h1[tid] = a * s + (be[tid] - mm[tid] * s);
  __syncthreads();
  if (tid < 32) {
    float a2 = b2[tid];
    for (int k = 0; k < 128; ++k) a2 += h1[k] * w2[k * 32 + tid];
    out[row * 32 + tid] = a2;
  }
}

// ---------------- edge predictor: [128x256]@[256x128] -> relu BN -> @[128x16] ----------------
__global__ __launch_bounds__(256) void k_edge_pred(
    const float* __restrict__ b1, const float* __restrict__ b2,
    const float* __restrict__ g, const float* __restrict__ be,
    const float* __restrict__ mm, const float* __restrict__ vv,
    const _Float16* __restrict__ w1t,  // [128][256] f16 (transposed)
    const _Float16* __restrict__ w2t,  // [16][128] f16 (transposed)
    const _Float16* __restrict__ eeff, const _Float16* __restrict__ eenc,
    float* __restrict__ out) {
  __shared__ _Float16 sAe[128 * 128];  // 32 KB (edge_effect tile)
  __shared__ _Float16 sAn[128 * 128];  // 32 KB (edge_enc tile)
  __shared__ _Float16 sB[128 * 128];   // 32 KB (w1 chunk, ldk=128)
  __shared__ _Float16 sH[128 * 128];   // 32 KB
  __shared__ _Float16 sW2[16 * 128];   // 4 KB
  __shared__ float sS[128], sT[128];
  int blk = blockIdx.x, tid = threadIdx.x;
  if (tid < 128) { float s = g[tid] * rsqrtf(vv[tid] + 1e-5f); sS[tid] = s; sT[tid] = be[tid] - mm[tid] * s; }
#if HAVE_TDM
  if (tid < 32) {
    tdm_load_2d((u32)(uintptr_t)sAe, eeff + (size_t)blk * 128 * 128, 128, 128, 128);
    tdm_load_2d((u32)(uintptr_t)sAn, eenc + (size_t)blk * 128 * 128, 128, 128, 128);
    tdm_load_2d((u32)(uintptr_t)sW2, w2t, 128, 16, 128);
  }
#else
  for (int i = tid; i < 2048; i += 256) {
    ((uint4*)sAe)[i] = ((const uint4*)(eeff + (size_t)blk * 128 * 128))[i];
    ((uint4*)sAn)[i] = ((const uint4*)(eenc + (size_t)blk * 128 * 128))[i];
  }
  for (int i = tid; i < 16 * 128 / 8; i += 256) ((uint4*)sW2)[i] = ((const uint4*)w2t)[i];
#endif
  int wave = tid >> 5, rowbase = wave * 16, lane = tid & 31;
  int colb = lane & 15, rhalf = (lane >> 4) * 8;
  v8f acc[8];
#pragma unroll
  for (int i = 0; i < 8; ++i) acc[i] = vzero();
  for (int half = 0; half < 2; ++half) {
    __syncthreads();
#if HAVE_TDM
    if (tid < 32) {
      tdm_load_2d((u32)(uintptr_t)sB, w1t + half * 128, 128, 128, 256);
      __builtin_amdgcn_s_wait_tensorcnt(0);
    }
#else
    for (int i = tid; i < 2048; i += 256) {
      int col = i >> 4, j = i & 15;
      ((uint4*)(sB + col * 128))[j] = ((const uint4*)(w1t + (size_t)col * 256 + half * 128))[j];
    }
#endif
    __syncthreads();
    const _Float16* sAh = half ? sAn : sAe;
#pragma unroll
    for (int ks = 0; ks < 4; ++ks) {
      v16h a = frag_a(sAh + rowbase * 128 + ks * 32, 128);
#pragma unroll
      for (int ht = 0; ht < 8; ++ht)
        acc[ht] = wmma_f16(a, frag_bt(sB + (ht * 16) * 128 + ks * 32, 128), acc[ht]);
    }
  }
#pragma unroll
  for (int ht = 0; ht < 8; ++ht) {
    int h = ht * 16 + colb;
#pragma unroll
    for (int r = 0; r < 8; ++r)
      sH[(rowbase + rhalf + r) * 128 + h] =
          (_Float16)(fmaxf(acc[ht][r] + b1[h], 0.f) * sS[h] + sT[h]);
  }
  __syncthreads();
  v8f c2 = vzero();
#pragma unroll
  for (int ks = 0; ks < 4; ++ks) {
    v16h a = frag_a(sH + rowbase * 128 + ks * 32, 128);
    c2 = wmma_f16(a, frag_bt(sW2 + ks * 32, 128), c2);
  }
#pragma unroll
  for (int r = 0; r < 8; ++r) {
    int row = rowbase + rhalf + r;
    out[((size_t)blk * 128 + row) * 16 + colb] = c2[r] + b2[colb];
  }
}

// ---------------- host side ----------------
extern "C" void kernel_launch(void* const* d_in, const int* in_sizes, int n_in,
                              void* d_out, int out_size, void* d_ws, size_t ws_size,
                              hipStream_t stream) {
  (void)in_sizes; (void)n_in; (void)out_size; (void)ws_size;
  const float* node_rep  = (const float*)d_in[0];
  const float* edge_rep  = (const float*)d_in[1];
  const float* edge_type = (const float*)d_in[2];
  const float* ee_b1 = (const float*)d_in[3];   // edge_enc
  const float* ee_be = (const float*)d_in[4];
  const float* ee_g  = (const float*)d_in[5];
  const float* ee_m  = (const float*)d_in[6];
  const float* ee_v  = (const float*)d_in[7];
  const float* ee_w1 = (const float*)d_in[8];
  const float* ep_b1 = (const float*)d_in[9];   // edge_pred
  const float* ep_b2 = (const float*)d_in[10];
  const float* ep_be = (const float*)d_in[11];
  const float* ep_g  = (const float*)d_in[12];
  const float* ep_m  = (const float*)d_in[13];
  const float* ep_v  = (const float*)d_in[14];
  const float* ep_w1 = (const float*)d_in[15];
  const float* ep_w2 = (const float*)d_in[16];
  const float* pp_b1[2], *pp_b2[2], *pp_be[2], *pp_g[2], *pp_m[2], *pp_v[2], *pp_w1[2], *pp_w2[2];
  for (int s = 0; s < 2; ++s) {
    int o = 17 + s * 8;
    pp_b1[s] = (const float*)d_in[o + 0]; pp_b2[s] = (const float*)d_in[o + 1];
    pp_be[s] = (const float*)d_in[o + 2]; pp_g[s]  = (const float*)d_in[o + 3];
    pp_m[s]  = (const float*)d_in[o + 4]; pp_v[s]  = (const float*)d_in[o + 5];
    pp_w1[s] = (const float*)d_in[o + 6]; pp_w2[s] = (const float*)d_in[o + 7];
  }
  const float* ne_b  = (const float*)d_in[33];  // node_enc
  const float* ne_be = (const float*)d_in[34];
  const float* ne_g  = (const float*)d_in[35];
  const float* ne_m  = (const float*)d_in[36];
  const float* ne_v  = (const float*)d_in[37];
  const float* ne_w  = (const float*)d_in[38];
  const float* nr_b1 = (const float*)d_in[39];  // node_pred
  const float* nr_b2 = (const float*)d_in[40];
  const float* nr_be = (const float*)d_in[41];
  const float* nr_g  = (const float*)d_in[42];
  const float* nr_m  = (const float*)d_in[43];
  const float* nr_v  = (const float*)d_in[44];
  const float* nr_w1 = (const float*)d_in[45];
  const float* nr_w2 = (const float*)d_in[46];
  const float* np_b1 = (const float*)d_in[47];  // node_prop
  const float* np_b2 = (const float*)d_in[48];
  const float* np_be = (const float*)d_in[49];
  const float* np_g  = (const float*)d_in[50];
  const float* np_m  = (const float*)d_in[51];
  const float* np_v  = (const float*)d_in[52];
  const float* np_w1 = (const float*)d_in[53];
  const float* np_w2 = (const float*)d_in[54];

  char* p = (char*)d_ws;
  auto take = [&](size_t bytes) { char* r = p; p += (bytes + 255) & ~(size_t)255; return r; };
  float*     nenc32 = (float*)take(1024 * 128 * 4);
  _Float16*  nenc16 = (_Float16*)take(1024 * 128 * 2);
  float*     neff32 = (float*)take(1024 * 128 * 4);
  _Float16*  neff16 = (_Float16*)take(1024 * 128 * 2);
  float*     eagg   = (float*)take(1024 * 128 * 4);
  _Float16*  eenc16 = (_Float16*)take((size_t)131072 * 128 * 2);
  _Float16*  eeff16 = (_Float16*)take((size_t)131072 * 128 * 2);
  _Float16*  w1t_enc = (_Float16*)take(256 * 96 * 2);
  _Float16*  w1t_p[2], *w2t_p[2];
  for (int s = 0; s < 2; ++s) {
    w1t_p[s] = (_Float16*)take(256 * 384 * 2);
    w2t_p[s] = (_Float16*)take(2 * 128 * 128 * 2);
  }
  _Float16*  wpred1t = (_Float16*)take(128 * 256 * 2);
  _Float16*  wpred2t = (_Float16*)take(16 * 128 * 2);

  // pack weights to transposed f16
  k_pack_eh_t<<<96, 256, 0, stream>>>(ee_w1, w1t_enc, 96);
  k_pack_eh_t<<<384, 256, 0, stream>>>(pp_w1[0], w1t_p[0], 384);
  k_pack_eh_t<<<384, 256, 0, stream>>>(pp_w1[1], w1t_p[1], 384);
  for (int s = 0; s < 2; ++s)
    for (int e = 0; e < 2; ++e)
      k_cvt_t<<<64, 256, 0, stream>>>(pp_w2[s] + e * 128 * 128, w2t_p[s] + e * 128 * 128, 128, 128);
  k_cvt_t<<<128, 256, 0, stream>>>(ep_w1, wpred1t, 256, 128);
  k_cvt_t<<<8, 256, 0, stream>>>(ep_w2, wpred2t, 128, 16);

  // encoders
  k_node_enc<<<1024, 128, 0, stream>>>(node_rep, ne_w, ne_b, ne_g, ne_be, ne_m, ne_v,
                                       nenc32, nenc16, neff32, neff16);
  k_edge_enc<<<1024, 256, 0, stream>>>(node_rep, edge_rep, edge_type,
                                       ee_b1, ee_g, ee_be, ee_m, ee_v, w1t_enc, eenc16);
  // propagation
  for (int s = 0; s < 2; ++s) {
    const _Float16* ein = (s == 0) ? eenc16 : eeff16;
    k_edge_prop<<<1024, 256, 0, stream>>>(edge_type,
                                          pp_b1[s], pp_b2[s], pp_g[s], pp_be[s], pp_m[s], pp_v[s],
                                          w1t_p[s], w2t_p[s],
                                          neff32, neff16, ein, eeff16, eagg);
    k_node_prop<<<1024, 128, 0, stream>>>(nenc32, neff32, eagg,
                                          np_w1, np_b1, np_w2, np_b2,
                                          np_g, np_be, np_m, np_v, neff32, neff16);
  }
  // predictors
  k_node_pred<<<1024, 128, 0, stream>>>(neff32, nenc32, nr_w1, nr_b1, nr_w2, nr_b2,
                                        nr_g, nr_be, nr_m, nr_v, (float*)d_out);
  k_edge_pred<<<1024, 256, 0, stream>>>(ep_b1, ep_b2, ep_g, ep_be, ep_m, ep_v,
                                        wpred1t, wpred2t, eeff16, eenc16,
                                        (float*)d_out + 32768);
}
